// ComplexModel_80204219285801
// MI455X (gfx1250) — compile-verified, matrix operand order checked by reading, not run on hardware
//
#include <hip/hip_runtime.h>

typedef __attribute__((ext_vector_type(16))) __bf16 v16bf;
typedef __attribute__((ext_vector_type(8)))  __bf16 v8bf;
typedef __attribute__((ext_vector_type(8)))  float  v8f;
typedef __attribute__((ext_vector_type(4)))  float  v4f;

#define T_ 1024

__device__ __forceinline__ float sigf(float x) { return 1.0f / (1.0f + __expf(-x)); }

// gfx1250 async DMA: copy 16 bytes global -> LDS without touching VGPR data path.
// Per-lane LDS byte address in %0, per-lane 64-bit global address in %1 (ASYNCcnt).
__device__ __forceinline__ void async_g2l_b128(unsigned lds_off, const void* gptr) {
  asm volatile("global_load_async_to_lds_b128 %0, %1, off"
               :: "v"(lds_off), "v"((unsigned long long)(uintptr_t)gptr)
               : "memory");
}
__device__ __forceinline__ void wait_async0() {
  asm volatile("s_wait_asynccnt 0x0" ::: "memory");
}

__device__ __forceinline__ void group_barrier(unsigned* ctr, unsigned target) {
  __threadfence();
  __syncthreads();
  if (threadIdx.x == 0) {
    __hip_atomic_fetch_add(ctr, 1u, __ATOMIC_ACQ_REL, __HIP_MEMORY_SCOPE_AGENT);
    while (__hip_atomic_load(ctr, __ATOMIC_ACQUIRE, __HIP_MEMORY_SCOPE_AGENT) < target) {
      __builtin_amdgcn_s_sleep(1);
    }
  }
  __syncthreads();
}

// ---------------------------------------------------------------------------
// Weight prep: bf16 concatenated weights Wcat[n][0:256]=Wih, [256:512]=Whh.
// LSTM rows gate-interleaved: column cidx = 4*j + gate (gate order i,f,g,o).
// Fuses bih+bhh; zeroes spin-barrier counters.
// ---------------------------------------------------------------------------
__global__ __launch_bounds__(256) void prep_kernel(
    const float* __restrict__ rWih, const float* __restrict__ rWhh,
    const float* __restrict__ rbih, const float* __restrict__ rbhh,
    const float* __restrict__ lWih, const float* __restrict__ lWhh,
    const float* __restrict__ lbih, const float* __restrict__ lbhh,
    __bf16* __restrict__ rnnW, __bf16* __restrict__ lstmW,
    float* __restrict__ rnnB, float* __restrict__ lstmB,
    unsigned* __restrict__ ctr)
{
  size_t i = (size_t)blockIdx.x * 256 + threadIdx.x;
  size_t stride = (size_t)gridDim.x * 256;
  if (i < 2) ctr[i] = 0u;

  for (size_t idx = i; idx < (size_t)2 * 256 * 512; idx += stride) {   // RNN Wcat
    size_t l = idx >> 17, rem = idx & 131071;
    size_t n = rem >> 9, k = rem & 511;
    float v = (k < 256) ? rWih[(l * 256 + n) * 256 + k]
                        : rWhh[(l * 256 + n) * 256 + (k - 256)];
    rnnW[idx] = (__bf16)v;
  }
  for (size_t idx = i; idx < (size_t)2 * 1024 * 512; idx += stride) {  // LSTM Wcat
    size_t l = idx >> 19, rem = idx & 524287;
    size_t cidx = rem >> 9, k = rem & 511;
    size_t orig = (cidx & 3) * 256 + (cidx >> 2);
    float v = (k < 256) ? lWih[(l * 1024 + orig) * 256 + k]
                        : lWhh[(l * 1024 + orig) * 256 + (k - 256)];
    lstmW[idx] = (__bf16)v;
  }
  for (size_t idx = i; idx < 512; idx += stride) rnnB[idx] = rbih[idx] + rbhh[idx];
  for (size_t idx = i; idx < 2048; idx += stride) {
    size_t l = idx >> 10, cidx = idx & 1023;
    size_t orig = (cidx & 3) * 256 + (cidx >> 2);
    lstmB[idx] = lbih[l * 1024 + orig] + lbhh[l * 1024 + orig];
  }
}

// ---------------------------------------------------------------------------
// Fused persistent scan. Blocks 0..7: RNN stack, blocks 8..39: LSTM stack,
// running concurrently with independent spin-barrier counters.
// Per step: stage A=[x_t | h_{t-1}] (64x512 bf16) into XOR-swizzled LDS
// (bf16 halves via global_load_async_to_lds_b128), each wave computes one
// 16x16 tile with 16 chained v_wmma_f32_16x16x32_bf16 (B fragments held in
// VGPRs for the whole layer), nonlinearity, write h, group barrier.
// ---------------------------------------------------------------------------
__global__ __launch_bounds__(256, 1) void fused_scan_kernel(
    const float* __restrict__ rnn_x, const float* __restrict__ lstm_x,
    const __bf16* __restrict__ rnnW, const __bf16* __restrict__ lstmW,
    const float* __restrict__ rnnB, const float* __restrict__ lstmB,
    __bf16* __restrict__ y_rnn, __bf16* __restrict__ y_lstm,
    __bf16* __restrict__ ring_rnn, __bf16* __restrict__ ring_lstm,
    float* __restrict__ feat, unsigned* __restrict__ ctr)
{
  __shared__ __bf16 sA[64 * 512];   // 64 KB, XOR-swizzled in 8-element chunks
  const unsigned sAbase = (unsigned)(uintptr_t)(void*)sA;  // LDS byte offset

  const int tid    = threadIdx.x;
  const int wave   = tid >> 5;
  const int lane   = tid & 31;
  const int lane15 = lane & 15;
  const int khalf  = (lane & 16) ? 8 : 0;   // 16-bit WMMA A/B layout K-half
  const int cIdx0  = khalf >> 3;

  const bool isL = (blockIdx.x >= 8);
  const int  blk = isL ? (int)blockIdx.x - 8 : (int)blockIdx.x;
  const unsigned gsize = isL ? 32u : 8u;
  unsigned* myctr = ctr + (isL ? 1 : 0);

  const int wg      = blk * 8 + wave;   // RNN: 64 tiles, LSTM: 256 tiles
  const int mtile   = wg & 3;           // 4 M-tiles over B=64
  const int ntile_g = wg >> 2;          // RNN: [0,16), LSTM: [0,64)

  const int aRowIdx = mtile * 16 + lane15;
  const __bf16* aRow = sA + (size_t)aRowIdx * 512;
  const int bswz = aRowIdx & 15;

  float creg[8];   // LSTM cell state lives in registers across all timesteps

  for (int layer = 0; layer < 2; ++layer) {
    const __bf16* Wc = isL ? (lstmW + (size_t)layer * 1024 * 512)
                           : (rnnW  + (size_t)layer * 256  * 512);
    const float*  Bc = isL ? (lstmB + layer * 1024) : (rnnB + layer * 256);
    const float bias_v = Bc[ntile_g * 16 + lane15];
    const __bf16* bRow = Wc + (size_t)(ntile_g * 16 + lane15) * 512 + khalf;

    // B fragments for the whole layer stay resident in VGPRs (16 x v16bf).
    v16bf bfrag[16];
    #pragma unroll
    for (int kk = 0; kk < 16; ++kk) {
      v8bf blo = *(const v8bf*)&bRow[kk * 32];
      v8bf bhi = *(const v8bf*)&bRow[kk * 32 + 16];
      bfrag[kk] = __builtin_shufflevector(blo, bhi,
                    0,1,2,3,4,5,6,7,8,9,10,11,12,13,14,15);
    }

    const float*  x32 = nullptr;
    const __bf16* x16 = nullptr;
    if (layer == 0) x32 = isL ? lstm_x : rnn_x;
    else            x16 = isL ? y_lstm : y_rnn;
    __bf16* yseq = isL ? y_lstm : y_rnn;
    __bf16* ring = isL ? ring_lstm : ring_rnn;

    #pragma unroll
    for (int r = 0; r < 8; ++r) creg[r] = 0.0f;

    for (int t = 0; t < T_; ++t) {
      // ---- stage A = [x_t | h_{t-1}] -> LDS bf16 (swizzled) ----
      const __bf16* hsrc =
          (t == 0) ? nullptr
                   : (layer == 0 ? (yseq + (size_t)(t - 1) * (64 * 256))
                                 : (ring + (size_t)((t - 1) & 1) * (64 * 256)));
      for (int idx = tid; idx < 64 * 64; idx += 256) {
        const int b  = idx >> 6;
        const int ck = idx & 63;       // chunk of 8 bf16
        const int k  = ck << 3;
        const unsigned ldsoff =
            sAbase + ((((unsigned)b << 9) + ((unsigned)(ck ^ (b & 15)) << 3)) << 1);
        if (k < 256) {
          if (layer == 0) {            // f32 -> bf16 convert on VALU path
            const float* xp = x32 + ((size_t)b * T_ + t) * 256 + k;
            v4f f0 = *(const v4f*)xp;
            v4f f1 = *(const v4f*)(xp + 4);
            v8bf v;
            v[0] = (__bf16)f0.x; v[1] = (__bf16)f0.y; v[2] = (__bf16)f0.z; v[3] = (__bf16)f0.w;
            v[4] = (__bf16)f1.x; v[5] = (__bf16)f1.y; v[6] = (__bf16)f1.z; v[7] = (__bf16)f1.w;
            *(v8bf*)&sA[((size_t)b << 9) + ((unsigned)(ck ^ (b & 15)) << 3)] = v;
            if (t + 1 < T_) __builtin_prefetch(xp + 256, 0, 0);   // next timestep x
          } else {                     // bf16: async DMA global -> LDS
            const __bf16* xp = x16 + ((size_t)t * 64 + b) * 256 + k;
            async_g2l_b128(ldsoff, xp);
            if (t + 1 < T_) __builtin_prefetch(xp + 64 * 256, 0, 0);
          }
        } else {
          if (t == 0) {
            v8bf z;
            #pragma unroll
            for (int e = 0; e < 8; ++e) z[e] = (__bf16)0.0f;
            *(v8bf*)&sA[((size_t)b << 9) + ((unsigned)(ck ^ (b & 15)) << 3)] = z;
          } else {                     // h_{t-1}: async DMA global -> LDS
            async_g2l_b128(ldsoff, hsrc + (size_t)b * 256 + (k - 256));
          }
        }
      }
      wait_async0();
      __syncthreads();

      // ---- one 16x16 tile: 16 chained WMMAs over K=512, A double-buffered ----
      v8f acc;
      #pragma unroll
      for (int r = 0; r < 8; ++r) acc[r] = bias_v;

      v8bf alo = *(const v8bf*)&aRow[(unsigned)((cIdx0    ) ^ bswz) << 3];
      v8bf ahi = *(const v8bf*)&aRow[(unsigned)((cIdx0 + 2) ^ bswz) << 3];
      #pragma unroll
      for (int kk = 0; kk < 16; ++kk) {
        v8bf nlo = alo, nhi = ahi;
        if (kk < 15) {                 // prefetch next A fragment before WMMA
          const int c0 = ((kk + 1) << 2) + cIdx0;
          nlo = *(const v8bf*)&aRow[(unsigned)( c0      ^ bswz) << 3];
          nhi = *(const v8bf*)&aRow[(unsigned)((c0 + 2) ^ bswz) << 3];
        }
        v16bf a = __builtin_shufflevector(alo, ahi,
                    0,1,2,3,4,5,6,7,8,9,10,11,12,13,14,15);
        acc = __builtin_amdgcn_wmma_f32_16x16x32_bf16(false, a, false, bfrag[kk],
                                                      (short)0, acc, false, false);
        alo = nlo; ahi = nhi;
      }

      // ---- epilogue: nonlinearity + write h ----
      __bf16* hdst = (layer == 0) ? (yseq + (size_t)t * (64 * 256))
                                  : (ring + (size_t)(t & 1) * (64 * 256));
      const bool lastFeat = (layer == 1) && (t == T_ - 1);
      if (!isL) {
        #pragma unroll
        for (int r = 0; r < 8; ++r) {
          float hv = tanhf(acc[r]);
          int m = mtile * 16 + r + ((lane >> 4) << 3);
          int n = ntile_g * 16 + lane15;
          hdst[(size_t)m * 256 + n] = (__bf16)hv;
          if (lastFeat) feat[(size_t)m * 512 + n] = hv;
        }
      } else {
        const int jl   = lane15 >> 2;          // hidden unit within tile (0..3)
        const int gsel = lane15 & 3;           // gate id (i,f,g,o)
        const int base = (lane & 16) | (jl << 2);
        #pragma unroll
        for (int r = 0; r < 8; ++r) {
          float gv = acc[r];
          float xi = __shfl(gv, base + 0, 32);
          float xf = __shfl(gv, base + 1, 32);
          float xg = __shfl(gv, base + 2, 32);
          float xo = __shfl(gv, base + 3, 32);
          float c  = sigf(xf) * creg[r] + sigf(xi) * tanhf(xg);
          creg[r] = c;
          float hv = sigf(xo) * tanhf(c);
          if (gsel == 0) {
            int m = mtile * 16 + r + ((lane >> 4) << 3);
            int j = ntile_g * 4 + jl;
            hdst[(size_t)m * 256 + j] = (__bf16)hv;
            if (lastFeat) feat[(size_t)m * 512 + 256 + j] = hv;
          }
        }
      }

      group_barrier(myctr, (unsigned)(layer * T_ + t + 1) * gsize);
    }
  }
}

// ---------------------------------------------------------------------------
// Final FC head: out[64,128] = feat[64,512] @ fc_W^T + fc_b   (8.4 MFLOP)
// ---------------------------------------------------------------------------
__global__ __launch_bounds__(256) void fc_kernel(
    const float* __restrict__ feat, const float* __restrict__ W,
    const float* __restrict__ b, float* __restrict__ out)
{
  int o = blockIdx.x * 256 + threadIdx.x;
  if (o >= 64 * 128) return;
  int m = o >> 7, c = o & 127;
  const float* fr = feat + (size_t)m * 512;
  const float* wr = W + (size_t)c * 512;
  float s = b[c];
  for (int k = 0; k < 512; k += 4) {
    v4f f = *(const v4f*)(fr + k);
    v4f g = *(const v4f*)(wr + k);
    s += f.x * g.x + f.y * g.y + f.z * g.z + f.w * g.w;
  }
  out[o] = s;
}

extern "C" void kernel_launch(void* const* d_in, const int* in_sizes, int n_in,
                              void* d_out, int out_size, void* d_ws, size_t ws_size,
                              hipStream_t stream) {
  (void)in_sizes; (void)n_in; (void)out_size; (void)ws_size;
  const float* rnn_x  = (const float*)d_in[0];
  const float* lstm_x = (const float*)d_in[1];
  const float* rWih   = (const float*)d_in[2];
  const float* rWhh   = (const float*)d_in[3];
  const float* rbih   = (const float*)d_in[4];
  const float* rbhh   = (const float*)d_in[5];
  const float* lWih   = (const float*)d_in[6];
  const float* lWhh   = (const float*)d_in[7];
  const float* lbih   = (const float*)d_in[8];
  const float* lbhh   = (const float*)d_in[9];
  const float* fcW    = (const float*)d_in[10];
  const float* fcb    = (const float*)d_in[11];
  float* out = (float*)d_out;

  char* w = (char*)d_ws;
  size_t o = 0;
  auto take = [&](size_t bytes) { size_t r = o; o += (bytes + 255) & ~(size_t)255; return r; };
  size_t o_ctr   = take(256);
  size_t o_feat  = take((size_t)64 * 512 * 4);
  size_t o_rnnW  = take((size_t)2 * 256 * 512 * 2);
  size_t o_lstmW = take((size_t)2 * 1024 * 512 * 2);
  size_t o_rnnB  = take((size_t)512 * 4);
  size_t o_lstmB = take((size_t)2048 * 4);
  size_t o_yr    = take((size_t)T_ * 64 * 256 * 2);
  size_t o_yl    = take((size_t)T_ * 64 * 256 * 2);
  size_t o_rr    = take((size_t)2 * 64 * 256 * 2);
  size_t o_rl    = take((size_t)2 * 64 * 256 * 2);

  unsigned* ctr  = (unsigned*)(w + o_ctr);
  float*  feat   = (float*)(w + o_feat);
  __bf16* rnnW   = (__bf16*)(w + o_rnnW);
  __bf16* lstmW  = (__bf16*)(w + o_lstmW);
  float*  rnnB   = (float*)(w + o_rnnB);
  float*  lstmB  = (float*)(w + o_lstmB);
  __bf16* y_rnn  = (__bf16*)(w + o_yr);
  __bf16* y_lstm = (__bf16*)(w + o_yl);
  __bf16* rg_rnn = (__bf16*)(w + o_rr);
  __bf16* rg_lstm= (__bf16*)(w + o_rl);

  prep_kernel<<<512, 256, 0, stream>>>(rWih, rWhh, rbih, rbhh,
                                       lWih, lWhh, lbih, lbhh,
                                       rnnW, lstmW, rnnB, lstmB, ctr);
  fused_scan_kernel<<<40, 256, 0, stream>>>(rnn_x, lstm_x, rnnW, lstmW, rnnB, lstmB,
                                            y_rnn, y_lstm, rg_rnn, rg_lstm, feat, ctr);
  fc_kernel<<<32, 256, 0, stream>>>(feat, fcW, fcb, out);
}
